// CRF_8443905704625
// MI455X (gfx1250) — matrix operand order repaired and checked by reading, not compiled, not run on hardware
//
#include <hip/hip_runtime.h>

// CRF loss on MI455X (gfx1250): log-semiring matrix chain product evaluated as
// scaled f16 WMMA GEMMs. 256 chunks x 32 transition matrices folded in
// parallel (B operand exp(trans - colmax) is step-invariant and LDS-resident),
// then an 8-level pairwise tree, then a tiny finalize kernel.
// Round 2: linear-domain rescaling epilogue (no per-element log/exp round-trip
// in the hot loop) + software-pipelined B-fragment loads in the GEMM.

#define NTOK 256
#define TLEN 8192
#define CHUNK 32
#define NCHUNK 256

typedef _Float16 v16h __attribute__((ext_vector_type(16)));
typedef float    v8f  __attribute__((ext_vector_type(8)));
typedef unsigned int v4u __attribute__((ext_vector_type(4)));

union Frag16 { v16h v; v4u q[2]; };

// ---------------------------------------------------------------------------
// Per-wave GEMM: S = Ea(256x256 f16, row-major LDS) @ Eb(256x256 f16, LDS).
// Wave owns tile-row r (16 rows), tile-cols (8g..8g+7). 64 WMMAs per call.
// A frag: lane lo<16 -> row 16r+lo, K {0..7,16..23}; hi half -> K {8..15,24..31}.
// B frag: lane -> col lo of the tile, same K split as A (16 strided K rows per lane).
// Software pipeline: prefetch next B (and next A at K-tile boundary) before
// issuing the current WMMA so waits are dscnt<=2 instead of dscnt==0.
// ---------------------------------------------------------------------------
__device__ __forceinline__ void lse_gemm_wave(const _Float16* Ea, const _Float16* Eb,
                                              int lane, int r, int g, v8f acc[8]) {
  const int hi = lane >> 4;
  const int lo = lane & 15;
  const _Float16* aRow = Ea + ((16 * r + lo) << 8);
  const int a0 = hi ? 8 : 0, a1 = hi ? 24 : 16;

  Frag16 a, an, b, bn;
  a.q[0] = *(const v4u*)(aRow + a0);
  a.q[1] = *(const v4u*)(aRow + a1);
  {
    const _Float16* bp = Eb + ((hi << 3) << 8) + (g << 7) + lo;
    for (int s = 0; s < 8; ++s) {
      b.v[s] = bp[s << 8];
      b.v[8 + s] = bp[(16 + s) << 8];
    }
  }
#pragma unroll
  for (int kk = 0; kk < 8; ++kk) {
    if (kk < 7) {
      const _Float16* ap = aRow + ((kk + 1) << 5);
      an.q[0] = *(const v4u*)(ap + a0);
      an.q[1] = *(const v4u*)(ap + a1);
    }
#pragma unroll
    for (int jt = 0; jt < 8; ++jt) {
      const int nk = (jt == 7) ? kk + 1 : kk;
      const int nj = (jt == 7) ? 0 : jt + 1;
      if (!(kk == 7 && jt == 7)) {
        const _Float16* bp = Eb + (((nk << 5) + (hi << 3)) << 8) + (g << 7) + (nj << 4) + lo;
        for (int s = 0; s < 8; ++s) {
          bn.v[s] = bp[s << 8];
          bn.v[8 + s] = bp[(16 + s) << 8];
        }
      }
      acc[jt] = __builtin_amdgcn_wmma_f32_16x16x32_f16(
          false, a.v, false, b.v, (short)0, acc[jt], false, false);
      b = bn;
      if (jt == 7) a = an;
    }
  }
}

// ---------------------------------------------------------------------------
// Prep: Ebg = exp(trans - colmax(trans)) in f16 (fixed B operand), ctmax[j].
// ---------------------------------------------------------------------------
__global__ void crf_prep_kernel(const float* __restrict__ trans,
                                _Float16* __restrict__ Ebg,
                                float* __restrict__ ctmax) {
  const int j = threadIdx.x;  // 256 threads
  float m = -3.0e38f;
  for (int k = 0; k < NTOK; ++k) m = fmaxf(m, trans[k * NTOK + j]);
  ctmax[j] = m;
  for (int k = 0; k < NTOK; ++k)
    Ebg[k * NTOK + j] = (_Float16)__expf(trans[k * NTOK + j] - m);
}

// ---------------------------------------------------------------------------
// Chunk kernel: block c folds M_{t0}..M_{t1} (t0=32c+1, t1=min(32c+32,8191))
// into one 256x256 log-space product written to outP[c]. 1024 threads.
// LDS: Ea 128K | Eb 128K | emitS 32K | raS/ranew/ctmx/cb/winv 1K | pbuf 2K.
// ---------------------------------------------------------------------------
__global__ void crf_chunk_kernel(const float* __restrict__ emitG,
                                 const float* __restrict__ transG,
                                 const _Float16* __restrict__ EbG,
                                 const float* __restrict__ ctmaxG,
                                 float* __restrict__ outP) {
  extern __shared__ __align__(16) char smem[];
  _Float16* Ea     = (_Float16*)(smem);
  _Float16* Eb     = (_Float16*)(smem + 131072);
  float*    emitS  = (float*)(smem + 262144);
  float*    raS    = (float*)(smem + 294912);
  float*    ranew  = (float*)(smem + 295936);
  float*    ctmxS  = (float*)(smem + 296960);
  float*    cbArr  = (float*)(smem + 297984);
  float*    winv   = (float*)(smem + 299008);
  float*    pbuf   = (float*)(smem + 300032);  // 512 floats
  float*    cbmaxS = (float*)(smem + 302080);  // 1 float

  const int tid  = threadIdx.x;
  const int c    = blockIdx.x;
  const int t0   = CHUNK * c + 1;
  const int t1   = min(CHUNK * c + CHUNK, TLEN - 1);
  const int S    = t1 - t0;        // matmul steps after init
  const int lane = tid & 31, w = tid >> 5, r = w >> 1, g = w & 1;
  const int hi   = lane >> 4, lo = lane & 15;

  // Stage fixed B operand + ctmax + this chunk's emit rows.
  {
    const v4u* src = (const v4u*)EbG;
    v4u* dst = (v4u*)Eb;
    for (int idx = tid; idx < 8192; idx += 1024) dst[idx] = src[idx];
  }
  if (tid < NTOK) ctmxS[tid] = ctmaxG[tid];
  for (int idx = tid; idx < (S + 1) * NTOK; idx += 1024)
    emitS[idx] = emitG[(size_t)t0 * NTOK + idx];
  __syncthreads();

  // Init product P = M_{t0} in row-shifted exp form (Ea, raS).
  if (tid < NTOK) {
    const float* trow = transG + tid * NTOK;
    float m = -3.0e38f;
    for (int j = 0; j < NTOK; ++j) m = fmaxf(m, trow[j] + emitS[j]);
    raS[tid] = m;
    for (int j = 0; j < NTOK; ++j)
      Ea[tid * NTOK + j] = (_Float16)__expf(trow[j] + emitS[j] - m);
  }

  for (int s = 1; s <= S; ++s) {
    if (tid < NTOK) cbArr[tid] = ctmxS[tid] + emitS[s * NTOK + tid];
    __syncthreads();  // Ea (prev step / init) + cbArr visible

    // Block-wide cbmax (wave 0): exp-overflow safety for the linear rescale.
    if (tid < 32) {
      float m = -3.0e38f;
#pragma unroll
      for (int q = 0; q < 8; ++q) m = fmaxf(m, cbArr[tid * 8 + q]);
      for (int msk = 1; msk < 32; msk <<= 1) m = fmaxf(m, __shfl_xor(m, msk, 32));
      if (tid == 0) cbmaxS[0] = m;
    }
    __syncthreads();

    v8f acc[8];
#pragma unroll
    for (int jt = 0; jt < 8; ++jt)
      acc[jt] = v8f{0.f, 0.f, 0.f, 0.f, 0.f, 0.f, 0.f, 0.f};
    lse_gemm_wave(Ea, Eb, lane, r, g, acc);

    if (s == S) {
      // Final step: log-domain write-out of the chunk product.
      float rav[8];
#pragma unroll
      for (int v = 0; v < 8; ++v) rav[v] = raS[16 * r + 8 * hi + v];
      float* O = outP + (size_t)c * (NTOK * NTOK);
#pragma unroll
      for (int jt = 0; jt < 8; ++jt) {
        const int jj = (g << 7) + (jt << 4) + lo;
        const float cbj = cbArr[jj];
#pragma unroll
        for (int v = 0; v < 8; ++v)
          O[(16 * r + 8 * hi + v) * NTOK + jj] =
              rav[v] + cbj + __logf(fmaxf(acc[jt][v], 1e-38f));
      }
    } else {
      // Linear-domain rescale: Lv = S * w_j; Wmax_i = max_j Lv;
      // Ea' = Lv / Wmax_i; ra'_i = ra_i + cbmax + log(Wmax_i).
      const float cbm = cbmaxS[0];
      float lw[8];
#pragma unroll
      for (int jt = 0; jt < 8; ++jt)
        lw[jt] = __expf(cbArr[(g << 7) + (jt << 4) + lo] - cbm);
#pragma unroll
      for (int jt = 0; jt < 8; ++jt)
#pragma unroll
        for (int v = 0; v < 8; ++v) acc[jt][v] *= lw[jt];

      float rmx[8];
#pragma unroll
      for (int v = 0; v < 8; ++v) {
        float m = 0.f;
#pragma unroll
        for (int jt = 0; jt < 8; ++jt) m = fmaxf(m, acc[jt][v]);
        for (int msk = 1; msk < 16; msk <<= 1) m = fmaxf(m, __shfl_xor(m, msk, 32));
        rmx[v] = m;
      }
      if (lo == 0) {
#pragma unroll
        for (int v = 0; v < 8; ++v) pbuf[g * NTOK + 16 * r + 8 * hi + v] = rmx[v];
      }
      __syncthreads();
      if (tid < NTOK) {
        const float Wm = fmaxf(fmaxf(pbuf[tid], pbuf[NTOK + tid]), 1e-30f);
        winv[tid]  = 1.0f / Wm;
        ranew[tid] = raS[tid] + cbmaxS[0] + __logf(Wm);
      }
      __syncthreads();
#pragma unroll
      for (int v = 0; v < 8; ++v) {
        const int i = 16 * r + 8 * hi + v;
        const float wi = winv[i];
#pragma unroll
        for (int jt = 0; jt < 8; ++jt)
          Ea[i * NTOK + (g << 7) + (jt << 4) + lo] = (_Float16)(acc[jt][v] * wi);
      }
      if (tid < NTOK) raS[tid] = ranew[tid];
    }
  }
}

// ---------------------------------------------------------------------------
// Tree kernel: out[b] = in[2b] (x) in[2b+1]  (log-semiring matmul), 1024 thr.
// LDS: Ea 128K | Eb 128K | ra 1K | cb 1K | red 4K.
// ---------------------------------------------------------------------------
__global__ void crf_tree_kernel(const float* __restrict__ in,
                                float* __restrict__ out) {
  extern __shared__ __align__(16) char smem[];
  _Float16* Ea = (_Float16*)(smem);
  _Float16* Eb = (_Float16*)(smem + 131072);
  float*    ra = (float*)(smem + 262144);
  float*    cb = (float*)(smem + 263168);
  float*    red = (float*)(smem + 264192);  // 4*256 floats

  const int tid = threadIdx.x;
  const float* L = in + (size_t)(2 * blockIdx.x) * (NTOK * NTOK);
  const float* R = L + NTOK * NTOK;

  // Column max of R -> cb; Eb = exp(R - cb_j) (coalesced).
  {
    const int j = tid & 255, kq = tid >> 8;
    float m = -3.0e38f;
    for (int k = kq * 64; k < kq * 64 + 64; ++k) m = fmaxf(m, R[k * NTOK + j]);
    red[kq * NTOK + j] = m;
  }
  __syncthreads();
  if (tid < NTOK)
    cb[tid] = fmaxf(fmaxf(red[tid], red[256 + tid]), fmaxf(red[512 + tid], red[768 + tid]));
  __syncthreads();
  for (int idx = tid; idx < NTOK * NTOK; idx += 1024)
    Eb[idx] = (_Float16)__expf(R[idx] - cb[idx & 255]);

  // Row max of L -> ra; Ea = exp(L - ra_i).
  {
    const int i = tid & 255, jq = tid >> 8;
    const float* row = L + i * NTOK + jq * 64;
    float m = -3.0e38f;
    for (int j = 0; j < 64; ++j) m = fmaxf(m, row[j]);
    red[jq * NTOK + i] = m;
  }
  __syncthreads();
  if (tid < NTOK)
    ra[tid] = fmaxf(fmaxf(red[tid], red[256 + tid]), fmaxf(red[512 + tid], red[768 + tid]));
  __syncthreads();
  for (int idx = tid; idx < NTOK * NTOK; idx += 1024)
    Ea[idx] = (_Float16)__expf(L[idx] - ra[idx >> 8]);
  __syncthreads();

  const int lane = tid & 31, w = tid >> 5, r = w >> 1, g = w & 1;
  const int hi = lane >> 4, lo = lane & 15;
  v8f acc[8];
#pragma unroll
  for (int jt = 0; jt < 8; ++jt)
    acc[jt] = v8f{0.f, 0.f, 0.f, 0.f, 0.f, 0.f, 0.f, 0.f};
  lse_gemm_wave(Ea, Eb, lane, r, g, acc);

  float* O = out + (size_t)blockIdx.x * (NTOK * NTOK);
#pragma unroll
  for (int jt = 0; jt < 8; ++jt) {
    const int jj = (g << 7) + (jt << 4) + lo;
    const float cbj = cb[jj];
#pragma unroll
    for (int v = 0; v < 8; ++v) {
      const int i = 16 * r + 8 * hi + v;
      O[i * NTOK + jj] = ra[i] + cbj + __logf(fmaxf(acc[jt][v], 1e-38f));
    }
  }
}

// ---------------------------------------------------------------------------
// Finalize: logZ = lse(alpha0 (x) P), gold score, out = logZ - gold. 256 thr.
// ---------------------------------------------------------------------------
__global__ void crf_final_kernel(const float* __restrict__ emit,
                                 const int* __restrict__ y,
                                 const float* __restrict__ trans,
                                 const float* __restrict__ BOS,
                                 const float* __restrict__ EOS,
                                 const float* __restrict__ P,
                                 float* __restrict__ out) {
  __shared__ float alpha0[NTOK];
  __shared__ float af[NTOK];
  __shared__ float red[NTOK];
  const int tid = threadIdx.x;

  alpha0[tid] = BOS[tid] + emit[tid];
  __syncthreads();

  float m = -3.0e38f;
  for (int k = 0; k < NTOK; ++k) m = fmaxf(m, alpha0[k] + P[k * NTOK + tid]);
  float s = 0.f;
  for (int k = 0; k < NTOK; ++k) s += __expf(alpha0[k] + P[k * NTOK + tid] - m);
  af[tid] = m + __logf(s);
  __syncthreads();

  // logZ = logsumexp(af)
  red[tid] = af[tid];
  __syncthreads();
  for (int o = 128; o > 0; o >>= 1) {
    if (tid < o) red[tid] = fmaxf(red[tid], red[tid + o]);
    __syncthreads();
  }
  const float mz = red[0];
  __syncthreads();
  red[tid] = __expf(af[tid] - mz);
  __syncthreads();
  for (int o = 128; o > 0; o >>= 1) {
    if (tid < o) red[tid] += red[tid + o];
    __syncthreads();
  }
  const float logZ = mz + __logf(red[0]);
  __syncthreads();

  // Gold score body: sum over i<T-1 of trans[y_i][y_{i+1}] + emit[i][y_i].
  float gs = 0.f;
  for (int i = tid; i < TLEN - 1; i += NTOK) {
    const int yi = y[i], yn = y[i + 1];
    gs += trans[yi * NTOK + yn] + emit[(size_t)i * NTOK + yi];
  }
  red[tid] = gs;
  __syncthreads();
  for (int o = 128; o > 0; o >>= 1) {
    if (tid < o) red[tid] += red[tid + o];
    __syncthreads();
  }
  if (tid == 0) {
    const int yl = y[TLEN - 1];
    const float gold = red[0] + BOS[y[0]] + EOS[yl] +
                       emit[(size_t)(TLEN - 1) * NTOK + yl];
    out[0] = logZ - gold;
  }
}

// ---------------------------------------------------------------------------
extern "C" void kernel_launch(void* const* d_in, const int* in_sizes, int n_in,
                              void* d_out, int out_size, void* d_ws, size_t ws_size,
                              hipStream_t stream) {
  (void)in_sizes; (void)n_in; (void)out_size; (void)ws_size;
  const float* emit  = (const float*)d_in[0];
  const int*   y     = (const int*)d_in[1];
  const float* trans = (const float*)d_in[2];
  const float* BOS   = (const float*)d_in[3];
  const float* EOS   = (const float*)d_in[4];
  float* out = (float*)d_out;

  char* ws = (char*)d_ws;
  _Float16* Ebg  = (_Float16*)ws;                        // 131072 B
  float*    ctmx = (float*)(ws + 131072);                // 1024 B
  float*    RA   = (float*)(ws + 135168);                // 256 matrices (67 MB)
  float*    RB   = (float*)(ws + 135168 + 67108864);     // 128 matrices (33.5 MB)

  const size_t smemChunk = 302336;  // fits 320KB WGP LDS
  const size_t smemTree  = 268288;

  crf_prep_kernel<<<dim3(1), dim3(256), 0, stream>>>(trans, Ebg, ctmx);
  crf_chunk_kernel<<<dim3(NCHUNK), dim3(1024), smemChunk, stream>>>(
      emit, trans, Ebg, ctmx, RA);

  float* cur = RA;
  float* nxt = RB;
  for (int n = NCHUNK / 2; n >= 1; n >>= 1) {
    crf_tree_kernel<<<dim3(n), dim3(1024), smemTree, stream>>>(cur, nxt);
    float* t = cur; cur = nxt; nxt = t;
  }
  crf_final_kernel<<<dim3(1), dim3(256), 0, stream>>>(
      emit, y, trans, BOS, EOS, cur, out);
}
